// ConcealingModel_11656541241867
// MI455X (gfx1250) — compile-verified, matrix operand order checked by reading, not run on hardware
//
#include <hip/hip_runtime.h>
#include <math.h>

typedef _Float16 h16;
typedef __attribute__((ext_vector_type(16))) _Float16 v16h;
typedef __attribute__((ext_vector_type(8)))  _Float16 v8h;
typedef __attribute__((ext_vector_type(8)))  float    v8f;
typedef __attribute__((ext_vector_type(4)))  unsigned int u32x4;
typedef __attribute__((ext_vector_type(8)))  unsigned int u32x8;

#define LSTR 40   // LDS row stride in halves: 64B data + 16B pad (TDM pad_interval=3/pad_amount=3)

// gfx1250 async global->LDS copy of 16 bytes (per lane). VDST holds the LDS
// byte offset, VADDR the 64-bit global address (GV mode, saddr = off).
#define ASYNC_B128(ldsp, gp)                                              \
  asm volatile("global_load_async_to_lds_b128 %0, %1, off"                \
               :: "v"((unsigned)(unsigned long long)(ldsp)),              \
                  "v"((unsigned long long)(gp)) : "memory")

template<int N> static __device__ __forceinline__ void wait_async_le() {
#if __has_builtin(__builtin_amdgcn_s_wait_asynccnt)
  __builtin_amdgcn_s_wait_asynccnt(N);
#else
  asm volatile("s_wait_asynccnt %0" :: "i"(N) : "memory");
#endif
}
template<int N> static __device__ __forceinline__ void wait_tensor_le() {
#if __has_builtin(__builtin_amdgcn_s_wait_tensorcnt)
  __builtin_amdgcn_s_wait_tensorcnt(N);
#else
  asm volatile("s_wait_tensorcnt %0" :: "i"(N) : "memory");
#endif
}

// Tensor Data Mover: load a (<=64 rows) x (32 halves) 2D tile into LDS with
// hardware zero-fill past (td0, td1) and LDS padding of 4 DWORDs per 16 DWORDs
// (-> rows land at LSTR*2 = 80B stride).  D# per CDNA5 ISA.
static __device__ __forceinline__ void tdm_load_tile(
    unsigned lds_off, unsigned long long gaddr,
    unsigned td0, unsigned td1, unsigned stride_el)
{
  u32x4 g0 = { 1u,                                     // count=1 user descriptor
               lds_off,                                // lds_addr (bytes)
               (unsigned)gaddr,                        // global_addr[31:0]
               (unsigned)((gaddr >> 32) & 0x01ffffffu) | (2u << 30) }; // [56:32] | type=2
  u32x8 g1 = { (1u << 16) | (1u << 20) | (3u << 22) | (3u << 25),
               //  data_size=2B | pad_enable | pad_interval=16dw | pad_amount=4dw
               (td0 & 0xffffu) << 16,                  // tensor_dim0[15:0]
               (td0 >> 16) | ((td1 & 0xffffu) << 16),  // tensor_dim0[31:16], tensor_dim1[15:0]
               (td1 >> 16) | (32u << 16),              // tensor_dim1[31:16], tile_dim0=32
               64u,                                    // tile_dim1=64, tile_dim2=0
               stride_el,                              // tensor_dim0_stride[31:0]
               0u, 0u };
  u32x4 g2 = { 1u, 1u, 0u, 0u };                       // tensor_dim2=1, tensor_dim3=1
  u32x4 g3 = { 0u, 1u << 16, 0u, 0u };                 // tensor_dim4=1
  asm volatile("tensor_load_to_lds %0, %1, %2, %3"
               :: "s"(g0), "s"(g1), "s"(g2), "s"(g3) : "memory");
}

// ---------------------------------------------------------------------------
// Implicit-GEMM conv, NHWC f16 in, weights pre-transposed f16 [Cout][ks*ks*Cin],
// f32 accumulate via v_wmma_f32_16x16x32_f16.  Block tile M=128 x N=64,
// 4 waves stacked in M, each wave owns 32x64 (2x4 WMMA accumulators).
// Double-buffered pipeline: A tile via global_load_async_to_lds_b128
// (ASYNCcnt, fixed 4 issues/wave/step + scalar fix-up for edge rows),
// B (weight) tile via the Tensor Data Mover (TENSORcnt, wave 0 issues).
// Supports stride, dilation, SAME padding, fused up2.
// ACT: 0=none 1=elu 2=relu.  OutT: _Float16 or float.
// ---------------------------------------------------------------------------
template<int ACT, typename OutT>
__global__ __launch_bounds__(128) void conv_wmma_kernel(
    const h16* __restrict__ in, const h16* __restrict__ wt,
    const float* __restrict__ bias, OutT* __restrict__ out,
    int Bn, int HinE, int WinE, int up, int Cin, int Cout,
    int Hout, int Wout, int ks, int stride, int rate, int pad)
{
  __shared__ h16 lA[2][128 * LSTR];
  __shared__ h16 lB[2][64 * LSTR];

  const int tid  = threadIdx.x;
  const int lane = tid & 31;
  const int wave = tid >> 5;

  const int m0 = blockIdx.x * 128;
  const int n0 = blockIdx.y * 64;
  const int M  = Bn * Hout * Wout;
  const int Hst = HinE >> up, Wst = WinE >> up;

  const int gm = m0 + tid;          // A staging: one thread per tile row
  int b = 0, oy = 0, ox = 0;
  if (gm < M) {
    b = gm / (Hout * Wout);
    int r2 = gm - b * Hout * Wout;
    oy = r2 / Wout; ox = r2 - oy * Wout;
  }

  const int  kcin   = (Cin + 31) >> 5;
  const int  ksteps = ks * ks * kcin;
  const long wrow   = (long)ks * ks * Cin;

  auto a_info = [&](int kk, const h16*& sreal, const h16*& sasync,
                    bool& rowok, bool& full, int& c0) {
    int tap = kk / kcin, ck = kk - tap * kcin;
    int ty = tap / ks, tx = tap - ty * ks;
    int iy = oy * stride + ty * rate - pad;
    int ix = ox * stride + tx * rate - pad;
    rowok = (gm < M) && (iy >= 0) && (iy < HinE) && (ix >= 0) && (ix < WinE);
    c0 = ck * 32;
    sreal = rowok ? in + (((long)b * Hst + (iy >> up)) * Wst + (ix >> up)) * Cin + c0 : in;
    full  = rowok && (c0 + 32 <= Cin);
    sasync = full ? sreal : in;     // aligned, safely readable dummy when not full
  };

  auto a_issue = [&](int kk, int buf) {   // branchless: always 4 async/wave/step
    const h16 *sr, *sa; bool rk, fl; int c0;
    a_info(kk, sr, sa, rk, fl, c0);
    h16* d = &lA[buf][tid * LSTR];
    ASYNC_B128(d,      sa);
    ASYNC_B128(d + 8,  sa + 8);
    ASYNC_B128(d + 16, sa + 16);
    ASYNC_B128(d + 24, sa + 24);
  };

  auto a_fixup = [&](int kk, int buf) {   // overwrite dummy/partial rows
    const h16 *sr, *sa; bool rk, fl; int c0;
    a_info(kk, sr, sa, rk, fl, c0);
    if (!fl) {
      h16* d = &lA[buf][tid * LSTR];
      #pragma unroll
      for (int j = 0; j < 32; ++j)
        d[j] = (rk && (c0 + j) < Cin) ? sr[j] : (h16)0.f;
    }
  };

  auto b_issue = [&](int kk, int buf) {   // one TDM DMA per block per k-step
    int tap = kk / kcin, ck = kk - tap * kcin;
    int c0 = ck * 32;
    tdm_load_tile((unsigned)(unsigned long long)(&lB[buf][0]),
                  (unsigned long long)(wt + (long)n0 * wrow + (long)tap * Cin + c0),
                  (unsigned)(Cin - c0), (unsigned)(Cout - n0), (unsigned)wrow);
  };

  v8f acc[2][4] = {};

  a_issue(0, 0);
  if (wave == 0) b_issue(0, 0);

  for (int kk = 0; kk < ksteps; ++kk) {
    const int cur = kk & 1;
    const bool more = (kk + 1 < ksteps);
    if (more) {
      a_issue(kk + 1, cur ^ 1);           // overlap next tile's DMA with compute
      if (wave == 0) b_issue(kk + 1, cur ^ 1);
    }
    if (more) { wait_async_le<4>(); wait_tensor_le<1>(); }   // in-order: cur landed
    else      { wait_async_le<0>(); wait_tensor_le<0>(); }
    a_fixup(kk, cur);
    __syncthreads();

    // ISA 16-bit A/B layout: lanes 0-15 hold K 0-7 & 16-23, lanes 16-31 hold K 8-15 & 24-31
    const int koff = (lane >> 4) * 8;
    v16h afr[2], bfr[4];
    #pragma unroll
    for (int i = 0; i < 2; ++i) {
      const int r = wave * 32 + i * 16 + (lane & 15);
      v8h alo = *(const v8h*)&lA[cur][r * LSTR + koff];
      v8h ahi = *(const v8h*)&lA[cur][r * LSTR + koff + 16];
      #pragma unroll
      for (int j = 0; j < 8; ++j) { afr[i][j] = alo[j]; afr[i][8 + j] = ahi[j]; }
    }
    #pragma unroll
    for (int i = 0; i < 4; ++i) {
      const int n = i * 16 + (lane & 15);
      v8h blo = *(const v8h*)&lB[cur][n * LSTR + koff];
      v8h bhi = *(const v8h*)&lB[cur][n * LSTR + koff + 16];
      #pragma unroll
      for (int j = 0; j < 8; ++j) { bfr[i][j] = blo[j]; bfr[i][8 + j] = bhi[j]; }
    }
    #pragma unroll
    for (int mi = 0; mi < 2; ++mi)
      #pragma unroll
      for (int ni = 0; ni < 4; ++ni)
        acc[mi][ni] = __builtin_amdgcn_wmma_f32_16x16x32_f16(
            false, afr[mi], false, bfr[ni], (short)0, acc[mi][ni], false, false);
    __syncthreads();
  }

  // Epilogue: C/D layout -> lane L, vgpr v: M = v + (L>=16)*8, N = L%16
  #pragma unroll
  for (int mi = 0; mi < 2; ++mi)
    #pragma unroll
    for (int ni = 0; ni < 4; ++ni)
      #pragma unroll
      for (int v = 0; v < 8; ++v) {
        const int m = m0 + wave * 32 + mi * 16 + v + ((lane >> 4) << 3);
        const int n = n0 + ni * 16 + (lane & 15);
        if (m < M && n < Cout) {
          float y = acc[mi][ni][v] + bias[n];
          if (ACT == 1) y = (y > 0.f) ? y : (__expf(y) - 1.f);
          else if (ACT == 2) y = fmaxf(y, 0.f);
          out[(long)m * Cout + n] = (OutT)y;
        }
      }
}

// ---------------------------------------------------------------------------
// Elementwise / helper kernels
// ---------------------------------------------------------------------------
__global__ void wcvt_kernel(const float* __restrict__ w, h16* __restrict__ wt,
                            int K, int Cout, long tot) {
  long i = (long)blockIdx.x * blockDim.x + threadIdx.x;
  if (i >= tot) return;
  int n = (int)(i / K), k = (int)(i - (long)n * K);
  wt[i] = (h16)w[(long)k * Cout + n];
}

__global__ void fill0_kernel(float* p, int n) {
  int i = blockIdx.x * blockDim.x + threadIdx.x;
  if (i < n) p[i] = 0.f;
}

__global__ void make_input_kernel(const float* __restrict__ x, const float* __restrict__ mask,
                                  h16* __restrict__ out, long tot) {
  long i = (long)blockIdx.x * blockDim.x + threadIdx.x;
  if (i >= tot) return;
  int c = (int)(i % 5);
  long bp = i / 5;                 // b*65536 + p
  int p = (int)(bp & 65535);
  float v;
  if (c < 3)      v = x[bp * 3 + c];
  else if (c == 3) v = 1.f;
  else             v = mask[p];
  out[i] = (h16)v;
}

__global__ void compose_kernel(const float* __restrict__ h3, const float* __restrict__ xin,
                               const float* __restrict__ mask, float* __restrict__ x1out,
                               h16* __restrict__ xnow, int tot) {
  int i = blockIdx.x * blockDim.x + threadIdx.x;   // b*65536 + p
  if (i >= tot) return;
  int p = i & 65535;
  float mk = mask[p];
  long b3 = (long)i * 3, b5 = (long)i * 5;
  #pragma unroll
  for (int c = 0; c < 3; ++c) {
    float v = fminf(fmaxf(h3[b3 + c], -1.f), 1.f);
    x1out[b3 + c] = v;
    float xc = v * mk + xin[b3 + c] * (1.f - mk);
    xnow[b5 + c] = (h16)xc;
  }
  xnow[b5 + 3] = (h16)1.f;
  xnow[b5 + 4] = (h16)mk;
}

__global__ void clip_kernel(const float* __restrict__ in, float* __restrict__ out, int n) {
  int i = blockIdx.x * blockDim.x + threadIdx.x;
  if (i < n) out[i] = fminf(fmaxf(in[i], -1.f), 1.f);
}

__global__ void concat_kernel(const h16* __restrict__ xh, const h16* __restrict__ h,
                              h16* __restrict__ out, long tot) {
  long i = (long)blockIdx.x * blockDim.x + threadIdx.x;
  if (i >= tot) return;
  int c = (int)(i & 255);
  long bp = i >> 8;                // b*4096 + p
  out[i] = (c < 128) ? xh[bp * 128 + c] : h[bp * 128 + (c - 128)];
}

// --------------------------- contextual attention ---------------------------
__global__ void att_down_kernel(const h16* __restrict__ in, h16* __restrict__ out, long tot) {
  long i = (long)blockIdx.x * blockDim.x + threadIdx.x;   // [b][py][px][c] 32x32x128
  if (i >= tot) return;
  int c = (int)(i & 127), px = (int)((i >> 7) & 31), py = (int)((i >> 12) & 31), b = (int)(i >> 17);
  out[i] = in[(((long)b * 64 + py * 2) * 64 + px * 2) * 128 + c];
}

__global__ void att_norm_kernel(const h16* __restrict__ fds, float* __restrict__ nrm, int tot) {
  int t = blockIdx.x * blockDim.x + threadIdx.x;          // b*1024 + l
  if (t >= tot) return;
  int b = t >> 10, l = t & 1023, py = l >> 5, px = l & 31;
  float s = 0.f;
  for (int ty = 0; ty < 3; ++ty)
    for (int tx = 0; tx < 3; ++tx) {
      int yy = py + ty - 1, xx = px + tx - 1;
      if (yy < 0 || yy >= 32 || xx < 0 || xx >= 32) continue;
      const h16* q = fds + (((long)b * 32 + yy) * 32 + xx) * 128;
      for (int c = 0; c < 128; ++c) { float v = (float)q[c]; s += v * v; }
    }
  nrm[t] = fmaxf(sqrtf(s), 1e-4f);
}

__global__ void att_wn_kernel(const h16* __restrict__ fds, const float* __restrict__ nrm,
                              h16* __restrict__ wn, long tot) {
  long i = (long)blockIdx.x * blockDim.x + threadIdx.x;   // [b][l][tap*128+c]
  if (i >= tot) return;
  int w1152 = (int)(i % 1152);
  long bl = i / 1152;
  int l = (int)(bl & 1023), b = (int)(bl >> 10);
  int tap = w1152 >> 7, c = w1152 & 127;
  int ty = tap / 3, tx = tap - ty * 3;
  int py = l >> 5, px = l & 31;
  int yy = py + ty - 1, xx = px + tx - 1;
  float v = 0.f;
  if (yy >= 0 && yy < 32 && xx >= 0 && xx < 32)
    v = (float)fds[(((long)b * 32 + yy) * 32 + xx) * 128 + c];
  wn[i] = (h16)(v / nrm[bl]);
}

__global__ void att_raw_kernel(const h16* __restrict__ bf, h16* __restrict__ raw, long tot) {
  long i = (long)blockIdx.x * blockDim.x + threadIdx.x;   // [b][l][(ky*4+kx)*128+c]
  if (i >= tot) return;
  int w2048 = (int)(i & 2047);
  int l = (int)((i >> 11) & 1023), b = (int)(i >> 21);
  int kk = w2048 >> 7, c = w2048 & 127;
  int ky = kk >> 2, kx = kk & 3;
  int py = l >> 5, px = l & 31;
  int yy = py * 2 + ky - 1, xx = px * 2 + kx - 1;
  h16 v = (h16)0.f;
  if (yy >= 0 && yy < 64 && xx >= 0 && xx < 64)
    v = bf[(((long)b * 64 + yy) * 64 + xx) * 128 + c];
  raw[i] = v;
}

__global__ void att_mm_kernel(const float* __restrict__ mask, float* __restrict__ mm) {
  int l = blockIdx.x * blockDim.x + threadIdx.x;
  if (l >= 1024) return;
  int py = l >> 5, px = l & 31;
  bool all0 = true;
  for (int ty = 0; ty < 3; ++ty)
    for (int tx = 0; tx < 3; ++tx) {
      int yy = py + ty - 1, xx = px + tx - 1;
      if (yy < 0 || yy >= 32 || xx < 0 || xx >= 32) continue;
      if (mask[(yy * 8) * 256 + xx * 8] != 0.f) all0 = false;
    }
  mm[l] = all0 ? 1.f : 0.f;
}

// diagonal identity-3x3 "fuse" conv over flattened (fh*fw, bh*bw); orient picks flattening
__global__ void fuse_kernel(const float* __restrict__ in, float* __restrict__ out,
                            int orient, long tot) {
  long i = (long)blockIdx.x * blockDim.x + threadIdx.x;
  if (i >= tot) return;
  int r = (int)(i & 1048575);
  long bbase = i - r;
  int bx = r & 31, by = (r >> 5) & 31, fx = (r >> 10) & 31, fy = (r >> 15) & 31;
  int u, v;
  if (orient == 0) { u = fy * 32 + fx; v = by * 32 + bx; }
  else             { u = fx * 32 + fy; v = bx * 32 + by; }
  float s = 0.f;
  #pragma unroll
  for (int d = -1; d <= 1; ++d) {
    int u2 = u + d, v2 = v + d;
    if (u2 < 0 || u2 >= 1024 || v2 < 0 || v2 >= 1024) continue;
    int fy2, fx2, by2, bx2;
    if (orient == 0) { fy2 = u2 >> 5; fx2 = u2 & 31; by2 = v2 >> 5; bx2 = v2 & 31; }
    else             { fx2 = u2 >> 5; fy2 = u2 & 31; bx2 = v2 >> 5; by2 = v2 & 31; }
    s += in[bbase + (((fy2 * 32 + fx2) * 32 + by2) * 32 + bx2)];
  }
  out[i] = s;
}

__global__ void softmax_kernel(const float* __restrict__ in, float* __restrict__ out,
                               const float* __restrict__ mm, float scale) {
  const int row = blockIdx.x;                 // b*1024 + (fy*32+fx)
  const float* r = in + (long)row * 1024;
  float* o = out + (long)row * 1024;
  __shared__ float red[256];
  int tid = threadIdx.x;
  float mx = -1e30f;
  for (int l = tid; l < 1024; l += 256) mx = fmaxf(mx, r[l] * mm[l] * scale);
  red[tid] = mx; __syncthreads();
  for (int s = 128; s > 0; s >>= 1) { if (tid < s) red[tid] = fmaxf(red[tid], red[tid + s]); __syncthreads(); }
  mx = red[0]; __syncthreads();
  float sum = 0.f;
  for (int l = tid; l < 1024; l += 256) sum += __expf(r[l] * mm[l] * scale - mx);
  red[tid] = sum; __syncthreads();
  for (int s = 128; s > 0; s >>= 1) { if (tid < s) red[tid] += red[tid + s]; __syncthreads(); }
  sum = red[0]; __syncthreads();
  float inv = 1.f / sum;
  for (int l = tid; l < 1024; l += 256)
    o[l] = __expf(r[l] * mm[l] * scale - mx) * inv * mm[l];
}

// VJP of strided conv (stride 2, k=4, SAME on 64): transposed conv with raw patches
__global__ void deconv_kernel(const float* __restrict__ yi, const h16* __restrict__ raw,
                              h16* __restrict__ out) {
  const int bix = blockIdx.x;                 // b*4096 + y*64 + x
  const int b = bix >> 12, r = bix & 4095, y = r >> 6, x = r & 63;
  const int c = threadIdx.x;
  float acc = 0.f;
  for (int ky = (y + 1) & 1; ky < 4; ky += 2) {
    int fy = (y + 1 - ky) >> 1;
    if (fy < 0 || fy >= 32) continue;
    for (int kx = (x + 1) & 1; kx < 4; kx += 2) {
      int fx = (x + 1 - kx) >> 1;
      if (fx < 0 || fx >= 32) continue;
      const float* yr = yi + (((long)b * 1024 + fy * 32 + fx) * 1024);
      const h16* rw = raw + ((long)b * 1024 * 2048) + (ky * 4 + kx) * 128 + c;
      for (int l = 0; l < 1024; ++l) acc += yr[l] * (float)rw[(long)l * 2048];
    }
  }
  out[(((long)b * 64 + y) * 64 + x) * 128 + c] = (h16)(acc * 0.25f);
}

// ---------------------------------------------------------------------------
// Host orchestration
// ---------------------------------------------------------------------------
struct LSpec { int cin, cout, ks; };
static const LSpec SP[42] = {
  {5,32,5},{32,64,3},{64,64,3},{64,128,3},{128,128,3},{128,128,3},{128,128,3},{128,128,3},
  {128,128,3},{128,128,3},{128,128,3},{128,128,3},{128,64,3},{64,64,3},{64,32,3},{32,16,3},{16,3,3},
  {5,32,5},{32,32,3},{32,64,3},{64,64,3},{64,128,3},{128,128,3},{128,128,3},{128,128,3},{128,128,3},{128,128,3},
  {5,32,5},{32,32,3},{32,64,3},{64,128,3},{128,128,3},{128,128,3},{128,128,3},{128,128,3},
  {256,128,3},{128,128,3},{128,64,3},{64,64,3},{64,32,3},{32,16,3},{16,3,3}
};
enum { IC1=0, IC2, IC3, IC4, IC5, IC6, IC7, IC8, IC9, IC10, IC11, IC12, IC13, IC14, IC15, IC16, IC17,
       IX1, IX2, IX3, IX4, IX5, IX6, IX7, IX8, IX9, IX10,
       IP1, IP2, IP3, IP4, IP5, IP6, IP9, IP10,
       IA11, IA12, IA13, IA14, IA15, IA16, IA17 };

static void launch_conv(hipStream_t st, const h16* in, const h16* wt, const float* bias,
                        void* out, bool f32o, int act, int Bn, int Hs, int Ws, int up,
                        int Cin, int Cout, int ks, int stride, int rate) {
  int HinE = Hs << up, WinE = Ws << up;
  int Hout = (HinE + stride - 1) / stride, Wout = (WinE + stride - 1) / stride;
  int padT = (Hout - 1) * stride + (ks - 1) * rate + 1 - HinE;
  int pad = padT > 0 ? padT / 2 : 0;
  int M = Bn * Hout * Wout;
  dim3 grid((M + 127) / 128, (Cout + 63) / 64);
  #define CONV_CASE(A, T) conv_wmma_kernel<A, T><<<grid, 128, 0, st>>>( \
      in, wt, bias, (T*)out, Bn, HinE, WinE, up, Cin, Cout, Hout, Wout, ks, stride, rate, pad)
  if (f32o) { if (act == 1) CONV_CASE(1, float); else if (act == 2) CONV_CASE(2, float); else CONV_CASE(0, float); }
  else      { if (act == 1) CONV_CASE(1, h16);   else if (act == 2) CONV_CASE(2, h16);   else CONV_CASE(0, h16);   }
  #undef CONV_CASE
}

extern "C" void kernel_launch(void* const* d_in, const int* in_sizes, int n_in,
                              void* d_out, int out_size, void* d_ws, size_t ws_size,
                              hipStream_t stream) {
  (void)in_sizes; (void)n_in; (void)out_size; (void)ws_size;
  const float* x    = (const float*)d_in[0];
  const float* mask = (const float*)d_in[1];

  // ---- workspace carve (bump allocator) ----
  char* base = (char*)d_ws; size_t off = 0;
  auto carve = [&](size_t bytes) -> void* {
    off = (off + 255) & ~(size_t)255; void* p = base + off; off += bytes; return p;
  };
  size_t wtoff[42]; size_t wtot = 0;
  for (int i = 0; i < 42; ++i) { wtoff[i] = wtot; wtot += (size_t)SP[i].ks * SP[i].ks * SP[i].cin * SP[i].cout; }
  h16*   WT   = (h16*)carve(wtot * 2);
  h16*   A    = (h16*)carve((size_t)8388608 * 2);   // ping
  h16*   Bb   = (h16*)carve((size_t)8388608 * 2);   // pong
  h16*   XNOW = (h16*)carve((size_t)1310720 * 2);
  h16*   XH   = (h16*)carve((size_t)2097152 * 2);
  float* T32  = (float*)carve((size_t)786432 * 4);
  h16*   FDS  = (h16*)carve((size_t)524288 * 2);
  h16*   WN   = (h16*)carve((size_t)4718592 * 2);
  h16*   RAW  = (h16*)carve((size_t)8388608 * 2);
  float* NRM  = (float*)carve((size_t)4096 * 4);
  float* MM   = (float*)carve((size_t)1024 * 4);
  float* ZB   = (float*)carve((size_t)1024 * 4);
  float* YI   = (float*)carve((size_t)4194304 * 4);
  float* YI2  = (float*)carve((size_t)4194304 * 4);

  // ---- convert all weights fp32 HWIO -> f16 [Cout][ks*ks*Cin] ----
  for (int i = 0; i < 42; ++i) {
    int K = SP[i].ks * SP[i].ks * SP[i].cin, Co = SP[i].cout;
    long tot = (long)K * Co;
    wcvt_kernel<<<(int)((tot + 255) / 256), 256, 0, stream>>>((const float*)d_in[2 + 2 * i], WT + wtoff[i], K, Co, tot);
  }
  fill0_kernel<<<4, 256, 0, stream>>>(ZB, 1024);

  auto conv = [&](int idx, const h16* pin, void* pout, int act, int Hs, int Ws,
                  int up, int stride, int rate, bool f32o) {
    launch_conv(stream, pin, WT + wtoff[idx], (const float*)d_in[3 + 2 * idx], pout, f32o, act,
                4, Hs, Ws, up, SP[idx].cin, SP[idx].cout, SP[idx].ks, stride, rate);
  };
  const int ELU = 1, RELU = 2, NONE = 0;

  // ---- stage 1: coarse network ----
  {
    long tot = (long)4 * 65536 * 5;
    make_input_kernel<<<(int)((tot + 255) / 256), 256, 0, stream>>>(x, mask, A, tot);
  }
  conv(IC1,  A,  Bb, ELU, 256, 256, 0, 1, 1, false);
  conv(IC2,  Bb, A,  ELU, 256, 256, 0, 2, 1, false);
  conv(IC3,  A,  Bb, ELU, 128, 128, 0, 1, 1, false);
  conv(IC4,  Bb, A,  ELU, 128, 128, 0, 2, 1, false);
  conv(IC5,  A,  Bb, ELU,  64,  64, 0, 1, 1, false);
  conv(IC6,  Bb, A,  ELU,  64,  64, 0, 1, 1, false);
  conv(IC7,  A,  Bb, ELU,  64,  64, 0, 1, 2, false);
  conv(IC8,  Bb, A,  ELU,  64,  64, 0, 1, 4, false);
  conv(IC9,  A,  Bb, ELU,  64,  64, 0, 1, 8, false);
  conv(IC10, Bb, A,  ELU,  64,  64, 0, 1, 16, false);
  conv(IC11, A,  Bb, ELU,  64,  64, 0, 1, 1, false);
  conv(IC12, Bb, A,  ELU,  64,  64, 0, 1, 1, false);
  conv(IC13, A,  Bb, ELU,  64,  64, 1, 1, 1, false);  // fused up2 -> 128 res
  conv(IC14, Bb, A,  ELU, 128, 128, 0, 1, 1, false);
  conv(IC15, A,  Bb, ELU, 128, 128, 1, 1, 1, false);  // fused up2 -> 256 res
  conv(IC16, Bb, A,  ELU, 256, 256, 0, 1, 1, false);
  conv(IC17, A,  T32, NONE, 256, 256, 0, 1, 1, true);
  compose_kernel<<<(4 * 65536 + 255) / 256, 256, 0, stream>>>(T32, x, mask, (float*)d_out, XNOW, 4 * 65536);

  // ---- stage 2: hallucination branch ----
  conv(IX1,  XNOW, A,  ELU, 256, 256, 0, 1, 1, false);
  conv(IX2,  A,  Bb, ELU, 256, 256, 0, 2, 1, false);
  conv(IX3,  Bb, A,  ELU, 128, 128, 0, 1, 1, false);
  conv(IX4,  A,  Bb, ELU, 128, 128, 0, 2, 1, false);
  conv(IX5,  Bb, A,  ELU,  64,  64, 0, 1, 1, false);
  conv(IX6,  A,  Bb, ELU,  64,  64, 0, 1, 1, false);
  conv(IX7,  Bb, A,  ELU,  64,  64, 0, 1, 2, false);
  conv(IX8,  A,  Bb, ELU,  64,  64, 0, 1, 4, false);
  conv(IX9,  Bb, A,  ELU,  64,  64, 0, 1, 8, false);
  conv(IX10, A,  XH, ELU,  64,  64, 0, 1, 16, false);

  // ---- stage 2: attention branch ----
  conv(IP1, XNOW, A,  ELU, 256, 256, 0, 1, 1, false);
  conv(IP2, A,  Bb, ELU, 256, 256, 0, 2, 1, false);
  conv(IP3, Bb, A,  ELU, 128, 128, 0, 1, 1, false);
  conv(IP4, A,  Bb, ELU, 128, 128, 0, 2, 1, false);
  conv(IP5, Bb, A,  ELU,  64,  64, 0, 1, 1, false);
  conv(IP6, A,  Bb, RELU, 64,  64, 0, 1, 1, false);   // f = b = Bb (4x64x64x128)

  { long t = (long)4 * 32 * 32 * 128;
    att_down_kernel<<<(int)((t + 255) / 256), 256, 0, stream>>>(Bb, FDS, t); }
  att_norm_kernel<<<(4096 + 127) / 128, 128, 0, stream>>>(FDS, NRM, 4096);
  { long t = (long)4 * 1024 * 1152;
    att_wn_kernel<<<(int)((t + 255) / 256), 256, 0, stream>>>(FDS, NRM, WN, t); }
  { long t = (long)4 * 1024 * 2048;
    att_raw_kernel<<<(int)((t + 255) / 256), 256, 0, stream>>>(Bb, RAW, t); }
  att_mm_kernel<<<4, 256, 0, stream>>>(mask, MM);

  // correlation = conv(fds, normalized patches) -> WMMA GEMM, per batch (Cout=1024)
  for (int b = 0; b < 4; ++b)
    launch_conv(stream, FDS + (long)b * 131072, WN + (long)b * 1179648, ZB,
                YI + (long)b * 1048576, true, 0, 1, 32, 32, 0, 128, 1024, 3, 1, 1);

  { long t = (long)4 << 20;
    fuse_kernel<<<(int)((t + 255) / 256), 256, 0, stream>>>(YI,  YI2, 0, t);
    fuse_kernel<<<(int)((t + 255) / 256), 256, 0, stream>>>(YI2, YI,  1, t); }
  softmax_kernel<<<4096, 256, 0, stream>>>(YI, YI2, MM, 10.f);
  deconv_kernel<<<4 * 4096, 128, 0, stream>>>(YI2, RAW, A);   // attention output -> A

  conv(IP9,  A,  Bb, ELU, 64, 64, 0, 1, 1, false);
  conv(IP10, Bb, A,  ELU, 64, 64, 0, 1, 1, false);
  { long t = (long)4 * 4096 * 256;
    concat_kernel<<<(int)((t + 255) / 256), 256, 0, stream>>>(XH, A, Bb, t); }

  conv(IA11, Bb, A,  ELU,  64,  64, 0, 1, 1, false);
  conv(IA12, A,  Bb, ELU,  64,  64, 0, 1, 1, false);
  conv(IA13, Bb, A,  ELU,  64,  64, 1, 1, 1, false);  // fused up2 -> 128
  conv(IA14, A,  Bb, ELU, 128, 128, 0, 1, 1, false);
  conv(IA15, Bb, A,  ELU, 128, 128, 1, 1, 1, false);  // fused up2 -> 256
  conv(IA16, A,  Bb, ELU, 256, 256, 0, 1, 1, false);
  conv(IA17, Bb, T32, NONE, 256, 256, 0, 1, 1, true);
  clip_kernel<<<(786432 + 255) / 256, 256, 0, stream>>>(T32, (float*)d_out + 786432, 786432);
}
